// FusingPartEncoders_34291018891297
// MI455X (gfx1250) — matrix operand order repaired and checked by reading, not compiled, not run on hardware
//
#include <hip/hip_runtime.h>
#include <stdint.h>

// ---------------------------------------------------------------------------
// Types for CDNA5 WMMA (wave32)
// ---------------------------------------------------------------------------
typedef __attribute__((ext_vector_type(16))) __bf16 v16bf;
typedef __attribute__((ext_vector_type(8)))  float  v8f;

union BF16Frag { v16bf v; unsigned short u[16]; };

__device__ __forceinline__ float bf2f(unsigned short h) {
  return __uint_as_float(((unsigned)h) << 16);
}
__device__ __forceinline__ unsigned short f2bf(float f) {
  unsigned u = __float_as_uint(f);
  unsigned r = (u >> 16) & 1u;              // round-to-nearest-even
  return (unsigned short)((u + 0x7fffu + r) >> 16);
}
// A-matrix (16x32 bf16) per-lane element->K map (ISA 7.12.2)
__device__ __forceinline__ int kmapA(int half, int e) {
  return (e < 8) ? (half * 8 + e) : (16 + half * 8 + (e - 8));
}
// B-matrix (32x16 bf16) per-lane element->K map (ISA 7.12.4/7.12.5 pattern)
__device__ __forceinline__ int kmapB(int half, int e) { return (e < 8) ? (half * 8 + e) : (16 + half * 8 + (e - 8)); }

// monotone float->int encoding (handles negatives) for atomic max
__device__ __forceinline__ int encf(float x) {
  int b = __float_as_int(x);
  return (b >= 0) ? b : (b ^ 0x7fffffff);
}
__device__ __forceinline__ float decf(int e) {
  int b = (e >= 0) ? e : (e ^ 0x7fffffff);
  return __int_as_float(b);
}

// ---------------------------------------------------------------------------
// Small utility kernels
// ---------------------------------------------------------------------------
__global__ void fill_f32_kernel(float* p, float v, long n) {
  long i = (long)blockIdx.x * blockDim.x + threadIdx.x;
  if (i < n) p[i] = v;
}
__global__ void fill_i32_kernel(int* p, int v, long n) {
  long i = (long)blockIdx.x * blockDim.x + threadIdx.x;
  if (i < n) p[i] = v;
}

// concat(pos[N,3], feat[N,61]) -> bf16 [N,64]
__global__ void build_xin_kernel(const float* __restrict__ pos,
                                 const float* __restrict__ feat,
                                 unsigned short* __restrict__ xin, int N) {
  long idx = (long)blockIdx.x * blockDim.x + threadIdx.x;
  if (idx >= (long)N * 64) return;
  int n = (int)(idx >> 6), c = (int)(idx & 63);
  float v = (c < 3) ? pos[(long)n * 3 + c] : feat[(long)n * 61 + (c - 3)];
  xin[idx] = f2bf(v);
}

// repack fp32 weights [K,C] into WMMA B-fragment order (bf16), optional per-K scale
__global__ void repack_w_kernel(const float* __restrict__ W,
                                const float* __restrict__ scale,
                                unsigned short* __restrict__ out, int K, int C) {
  long idx = (long)blockIdx.x * blockDim.x + threadIdx.x;
  long total = (long)K * C;
  if (idx >= total) return;
  int e    = (int)(idx & 15);
  int lane = (int)((idx >> 4) & 31);
  long frag = idx >> 9;
  int CT = C / 16;
  int kt = (int)(frag / CT), ct = (int)(frag % CT);
  int half = lane >> 4, n = lane & 15;
  int k = kt * 32 + kmapB(half, e);
  int c = ct * 16 + n;
  float w = W[(size_t)k * C + c];
  if (scale) w *= scale[k];
  out[idx] = f2bf(w);
}

// b2'[c] = b2[c] + sum_k t[k]*W2[k,c]   (BN folded into next linear)
__global__ void fold_bias_kernel(const float* __restrict__ b2,
                                 const float* __restrict__ W2,
                                 const float* __restrict__ t,
                                 float* __restrict__ out, int K, int C) {
  int c = blockIdx.x * blockDim.x + threadIdx.x;
  if (c >= C) return;
  float acc = b2[c];
  for (int k = 0; k < K; ++k) acc += t[k] * W2[(size_t)k * C + c];
  out[c] = acc;
}

// BN scale/shift from accumulated sum / sumsq
__global__ void stats_st_kernel(const float* __restrict__ sum,
                                const float* __restrict__ sq,
                                const float* __restrict__ g,
                                const float* __restrict__ be, float cnt,
                                float* __restrict__ sbuf, float* __restrict__ tbuf,
                                int C) {
  int c = blockIdx.x * blockDim.x + threadIdx.x;
  if (c >= C) return;
  float m = sum[c] / cnt;
  float v = sq[c] / cnt - m * m;
  if (v < 0.f) v = 0.f;
  float s = g[c] * rsqrtf(v + 1e-5f);
  sbuf[c] = s;
  tbuf[c] = be[c] - m * s;
}

// ---------------------------------------------------------------------------
// GEMM1 of a GCU layer: A built on the fly from gathered node rows
//   msg[e][k] = x[dst][k]            (k <  CIN)
//             = x[src][k-CIN] - x[dst][k-CIN]  (k >= CIN)
// D = relu(msg @ W1 + b1); store bf16 h1; accumulate per-channel stats.
// ---------------------------------------------------------------------------
template <int CIN, int COUT>
__global__ __launch_bounds__(128) void gemm_msg_kernel(
    const unsigned short* __restrict__ nodes, int ld, int colOff,
    const int* __restrict__ srcIdx, const int* __restrict__ dstIdx,
    const unsigned short* __restrict__ wrep, const float* __restrict__ bias,
    unsigned short* __restrict__ hout, float* __restrict__ gsum,
    float* __restrict__ gsq, int E) {
  constexpr int KT = (2 * CIN) / 32;
  constexpr int CT = COUT / 16;
  __shared__ float s_sum[COUT];
  __shared__ float s_sq[COUT];
  for (int c = threadIdx.x; c < COUT; c += blockDim.x) { s_sum[c] = 0.f; s_sq[c] = 0.f; }
  __syncthreads();

  const int wave = threadIdx.x >> 5;
  const int lane = threadIdx.x & 31;
  const int tile = blockIdx.x * (blockDim.x >> 5) + wave;
  const int rowBase = tile * 16;

  if (rowBase < E) {
    const int m = lane & 15;
    const int half = lane >> 4;
    const int e = min(rowBase + m, E - 1);
    const int sr = srcIdx[e];
    const int dr = dstIdx[e];
    const unsigned short* xs = nodes + (size_t)sr * ld + colOff;
    const unsigned short* xd = nodes + (size_t)dr * ld + colOff;

    const v8f vz = {0.f, 0.f, 0.f, 0.f, 0.f, 0.f, 0.f, 0.f};
    v8f acc[CT];
#pragma unroll
    for (int ct = 0; ct < CT; ++ct) acc[ct] = vz;

    for (int kt = 0; kt < KT; ++kt) {
      BF16Frag a;
      if ((kt + 1) * 32 <= CIN) {  // compile-time per kt (CIN multiple of 32)
#pragma unroll
        for (int i = 0; i < 16; ++i) a.u[i] = xd[kt * 32 + kmapA(half, i)];
      } else {
#pragma unroll
        for (int i = 0; i < 16; ++i) {
          int k = kt * 32 + kmapA(half, i) - CIN;
          a.u[i] = f2bf(bf2f(xs[k]) - bf2f(xd[k]));
        }
      }
#pragma unroll
      for (int ct = 0; ct < CT; ++ct) {
        BF16Frag b;
        const unsigned short* wp = wrep + ((size_t)(kt * CT + ct) << 9) + lane * 16;
#pragma unroll
        for (int i = 0; i < 16; ++i) b.u[i] = wp[i];
        acc[ct] = __builtin_amdgcn_wmma_f32_16x16x32_bf16(
            false, a.v, false, b.v, (short)0, acc[ct], false, false);
      }
    }

#pragma unroll
    for (int ct = 0; ct < CT; ++ct) {
      const int ch = ct * 16 + (lane & 15);
      const float bv = bias[ch];
      float ps = 0.f, pq = 0.f;
#pragma unroll
      for (int j = 0; j < 8; ++j) {
        const int row = rowBase + j + 8 * half;
        float v = acc[ct][j] + bv;
        v = v > 0.f ? v : 0.f;
        if (row < E) {
          hout[(size_t)row * COUT + ch] = f2bf(v);
          ps += v;
          pq += v * v;
        }
      }
      atomicAdd(&s_sum[ch], ps);
      atomicAdd(&s_sq[ch], pq);
    }
  }
  __syncthreads();
  for (int c = threadIdx.x; c < COUT; c += blockDim.x) {
    atomicAdd(&gsum[c], s_sum[c]);
    atomicAdd(&gsq[c], s_sq[c]);
  }
}

// ---------------------------------------------------------------------------
// Row-major bf16 GEMM: D = relu(A @ W + b).
// AGG=true : fuse segment-max (atomicMax of raw float bits; relu output >= 0)
// AGG=false: store bf16 output. Both accumulate per-channel stats.
// ---------------------------------------------------------------------------
template <int KK, int COUT, bool AGG>
__global__ __launch_bounds__(128) void gemm_rows_kernel(
    const unsigned short* __restrict__ Arows, int ld,
    const unsigned short* __restrict__ wrep, const float* __restrict__ bias,
    const int* __restrict__ dstIdx, int* __restrict__ agg,
    unsigned short* __restrict__ hout, float* __restrict__ gsum,
    float* __restrict__ gsq, int R) {
  constexpr int KT = KK / 32;
  constexpr int CT = COUT / 16;
  __shared__ float s_sum[COUT];
  __shared__ float s_sq[COUT];
  for (int c = threadIdx.x; c < COUT; c += blockDim.x) { s_sum[c] = 0.f; s_sq[c] = 0.f; }
  __syncthreads();

  const int wave = threadIdx.x >> 5;
  const int lane = threadIdx.x & 31;
  const int tile = blockIdx.x * (blockDim.x >> 5) + wave;
  const int rowBase = tile * 16;

  if (rowBase < R) {
    const int m = lane & 15;
    const int half = lane >> 4;
    const unsigned short* ap = Arows + (size_t)min(rowBase + m, R - 1) * ld;

    const v8f vz = {0.f, 0.f, 0.f, 0.f, 0.f, 0.f, 0.f, 0.f};
    v8f acc[CT];
#pragma unroll
    for (int ct = 0; ct < CT; ++ct) acc[ct] = vz;

    for (int kt = 0; kt < KT; ++kt) {
      BF16Frag a;
#pragma unroll
      for (int i = 0; i < 16; ++i) a.u[i] = ap[kt * 32 + kmapA(half, i)];
#pragma unroll
      for (int ct = 0; ct < CT; ++ct) {
        BF16Frag b;
        const unsigned short* wp = wrep + ((size_t)(kt * CT + ct) << 9) + lane * 16;
#pragma unroll
        for (int i = 0; i < 16; ++i) b.u[i] = wp[i];
        acc[ct] = __builtin_amdgcn_wmma_f32_16x16x32_bf16(
            false, a.v, false, b.v, (short)0, acc[ct], false, false);
      }
    }

    int drow[8];
    if constexpr (AGG) {
#pragma unroll
      for (int j = 0; j < 8; ++j) drow[j] = dstIdx[min(rowBase + j + 8 * half, R - 1)];
    }
#pragma unroll
    for (int ct = 0; ct < CT; ++ct) {
      const int ch = ct * 16 + (lane & 15);
      const float bv = bias[ch];
      float ps = 0.f, pq = 0.f;
#pragma unroll
      for (int j = 0; j < 8; ++j) {
        const int row = rowBase + j + 8 * half;
        float v = acc[ct][j] + bv;
        v = v > 0.f ? v : 0.f;
        if (row < R) {
          if constexpr (AGG) {
            atomicMax(&agg[(size_t)drow[j] * COUT + ch], __float_as_int(v));
          } else {
            hout[(size_t)row * COUT + ch] = f2bf(v);
          }
          ps += v;
          pq += v * v;
        }
      }
      atomicAdd(&s_sum[ch], ps);
      atomicAdd(&s_sq[ch], pq);
    }
  }
  __syncthreads();
  for (int c = threadIdx.x; c < COUT; c += blockDim.x) {
    atomicAdd(&gsum[c], s_sum[c]);
    atomicAdd(&gsq[c], s_sq[c]);
  }
}

// decode per-node max, apply BN2 affine, -inf sentinel -> 0, write x123 slice (bf16)
__global__ void finalize_nodes_kernel(const int* __restrict__ agg,
                                      const float* __restrict__ s,
                                      const float* __restrict__ t,
                                      unsigned short* __restrict__ x123bf, int N,
                                      int COUT, int colOff) {
  long idx = (long)blockIdx.x * blockDim.x + threadIdx.x;
  if (idx >= (long)N * COUT) return;
  int n = (int)(idx / COUT), c = (int)(idx % COUT);
  int e = agg[idx];
  float v = (e == (int)0xFF800000) ? 0.f : (s[c] * __int_as_float(e) + t[c]);
  x123bf[(size_t)n * 448 + colOff + c] = f2bf(v);
}

// segment-max of x123 over graphs (monotone int encoding handles negatives)
__global__ void segmax_kernel(const unsigned short* __restrict__ x123bf,
                              const int* __restrict__ batch,
                              int* __restrict__ xgenc, int N) {
  long idx = (long)blockIdx.x * blockDim.x + threadIdx.x;
  if (idx >= (long)N * 448) return;
  int n = (int)(idx / 448), c = (int)(idx % 448);
  float v = bf2f(x123bf[idx]);
  atomicMax(&xgenc[(size_t)batch[n] * 448 + c], encf(v));
}

__global__ void xg_decode_kernel(const int* __restrict__ xgenc,
                                 float* __restrict__ xgdec, int total) {
  int i = blockIdx.x * blockDim.x + threadIdx.x;
  if (i >= total) return;
  int e = xgenc[i];
  xgdec[i] = (e == (int)0x80000000) ? 0.f : decf(e);
}

// xg @ m2_w + b, relu  (8 graphs x 256 out)
__global__ void m2_kernel(const float* __restrict__ xgdec,
                          const float* __restrict__ w, const float* __restrict__ b,
                          float* __restrict__ h) {
  int g = blockIdx.x;
  int o = threadIdx.x;  // 256
  float acc = b[o];
  for (int k = 0; k < 448; ++k) acc += xgdec[g * 448 + k] * w[(size_t)k * 256 + o];
  h[g * 256 + o] = acc > 0.f ? acc : 0.f;
}

// BN over the 8 graph rows
__global__ void m2_bn_kernel(const float* __restrict__ h, const float* __restrict__ g,
                             const float* __restrict__ be, float* __restrict__ xgbn) {
  int o = blockIdx.x * blockDim.x + threadIdx.x;
  if (o >= 256) return;
  float s = 0.f, sq = 0.f;
  for (int r = 0; r < 8; ++r) { float v = h[r * 256 + o]; s += v; sq += v * v; }
  float m = s / 8.f, var = sq / 8.f - m * m;
  if (var < 0.f) var = 0.f;
  float sc = g[o] * rsqrtf(var + 1e-5f);
  float t = be[o] - m * sc;
  for (int r = 0; r < 8; ++r) xgbn[r * 256 + o] = sc * h[r * 256 + o] + t;
}

// x5 = concat(BN(h4), xg_bn[batch])   [N,512] fp32
__global__ void x5_kernel(const unsigned short* __restrict__ h4,
                          const float* __restrict__ s4, const float* __restrict__ t4,
                          const float* __restrict__ xgbn, const int* __restrict__ batch,
                          float* __restrict__ x5, int N) {
  long idx = (long)blockIdx.x * blockDim.x + threadIdx.x;
  if (idx >= (long)N * 512) return;
  int n = (int)(idx >> 9), c = (int)(idx & 511);
  float v;
  if (c < 256) v = s4[c] * bf2f(h4[(size_t)n * 256 + c]) + t4[c];
  else         v = xgbn[(size_t)batch[n] * 256 + (c - 256)];
  x5[idx] = v;
}

__global__ void ranges_init_kernel(int* gs, int* ge, int N) {
  int b = threadIdx.x;
  if (b < 8) { gs[b] = N; ge[b] = 0; }
}
__global__ void ranges_kernel(const int* __restrict__ batch, int* gs, int* ge, int N) {
  int n = blockIdx.x * blockDim.x + threadIdx.x;
  if (n >= N) return;
  int b = batch[n];
  atomicMin(&gs[b], n);
  atomicMax(&ge[b], n + 1);
}

// x6[b,k,c] = sum_{n in graph b} hm[n,k] * x5[n,c]     grid (8,4) x 128
__global__ void einsum_kernel(const float* __restrict__ x5,
                              const float* __restrict__ hm,
                              const int* __restrict__ gs, const int* __restrict__ ge,
                              float* __restrict__ x6raw) {
  __shared__ float s_hm[64 * 24];
  int b = blockIdx.x;
  int c = blockIdx.y * 128 + threadIdx.x;
  float acc[24];
#pragma unroll
  for (int k = 0; k < 24; ++k) acc[k] = 0.f;
  int n0 = gs[b], n1 = ge[b];
  for (int base = n0; base < n1; base += 64) {
    int cnt = min(64, n1 - base);
    __syncthreads();
    for (int i = threadIdx.x; i < cnt * 24; i += 128) s_hm[i] = hm[(size_t)base * 24 + i];
    __syncthreads();
    for (int nn = 0; nn < cnt; ++nn) {
      float xv = x5[(size_t)(base + nn) * 512 + c];
#pragma unroll
      for (int k = 0; k < 24; ++k) acc[k] += s_hm[nn * 24 + k] * xv;
    }
  }
#pragma unroll
  for (int k = 0; k < 24; ++k) x6raw[(size_t)(b * 24 + k) * 512 + c] = acc[k];
}

// h6 = relu(x6 @ lin_w + lin_b), accumulate final BN stats. grid 192 x 256
__global__ void lin_kernel(const float* __restrict__ x6raw, const float* __restrict__ w,
                           const float* __restrict__ b, float* __restrict__ h6,
                           float* __restrict__ fsum, float* __restrict__ fsq) {
  __shared__ float srow[512];
  int r = blockIdx.x;
  for (int i = threadIdx.x; i < 512; i += 256) srow[i] = x6raw[(size_t)r * 512 + i];
  __syncthreads();
  int o = threadIdx.x;
  float acc = b[o];
  for (int k = 0; k < 512; ++k) acc += srow[k] * w[(size_t)k * 256 + o];
  acc = acc > 0.f ? acc : 0.f;
  h6[(size_t)r * 256 + o] = acc;
  atomicAdd(&fsum[o], acc);
  atomicAdd(&fsq[o], acc * acc);
}

__global__ void final_kernel(const float* __restrict__ h6, const float* __restrict__ s,
                             const float* __restrict__ t, float* __restrict__ out) {
  int idx = blockIdx.x * blockDim.x + threadIdx.x;
  if (idx >= 192 * 256) return;
  int o = idx & 255;
  out[idx] = s[o] * h6[idx] + t[o];
}

// ---------------------------------------------------------------------------
// Host side
// ---------------------------------------------------------------------------
static inline int cdiv_l(long a, long b) { return (int)((a + b - 1) / b); }

template <int CIN, int COUT>
static void run_gcu(hipStream_t stream, const unsigned short* nodes, int ld, int colOff,
                    const int* src, const int* dst, int E, int N,
                    const float* w1, const float* b1, const float* g1, const float* be1,
                    const float* w2, const float* b2, const float* g2, const float* be2,
                    unsigned short* wrepA, unsigned short* wrepB, unsigned short* h1,
                    int* agg, float* gsum, float* gsq, float* sbuf, float* tbuf,
                    float* b2p, unsigned short* x123bf, int outColOff) {
  fill_f32_kernel<<<1, 256, 0, stream>>>(gsum, 0.f, (long)COUT);
  fill_f32_kernel<<<1, 256, 0, stream>>>(gsq, 0.f, (long)COUT);
  repack_w_kernel<<<cdiv_l((long)2 * CIN * COUT, 256), 256, 0, stream>>>(
      w1, nullptr, wrepA, 2 * CIN, COUT);
  const int blocksE = cdiv_l(cdiv_l(E, 16), 4);
  gemm_msg_kernel<CIN, COUT><<<blocksE, 128, 0, stream>>>(
      nodes, ld, colOff, src, dst, wrepA, b1, h1, gsum, gsq, E);
  stats_st_kernel<<<1, COUT, 0, stream>>>(gsum, gsq, g1, be1, (float)E, sbuf, tbuf, COUT);
  repack_w_kernel<<<cdiv_l((long)COUT * COUT, 256), 256, 0, stream>>>(
      w2, sbuf, wrepB, COUT, COUT);
  fold_bias_kernel<<<cdiv_l(COUT, 256), 256, 0, stream>>>(b2, w2, tbuf, b2p, COUT, COUT);
  fill_f32_kernel<<<1, 256, 0, stream>>>(gsum, 0.f, (long)COUT);
  fill_f32_kernel<<<1, 256, 0, stream>>>(gsq, 0.f, (long)COUT);
  fill_i32_kernel<<<cdiv_l((long)N * COUT, 256), 256, 0, stream>>>(
      agg, (int)0xFF800000, (long)N * COUT);
  gemm_rows_kernel<COUT, COUT, true><<<blocksE, 128, 0, stream>>>(
      h1, COUT, wrepB, b2p, dst, agg, nullptr, gsum, gsq, E);
  stats_st_kernel<<<1, COUT, 0, stream>>>(gsum, gsq, g2, be2, (float)E, sbuf, tbuf, COUT);
  finalize_nodes_kernel<<<cdiv_l((long)N * COUT, 256), 256, 0, stream>>>(
      agg, sbuf, tbuf, x123bf, N, COUT, outColOff);
}

extern "C" void kernel_launch(void* const* d_in, const int* in_sizes, int n_in,
                              void* d_out, int out_size, void* d_ws, size_t ws_size,
                              hipStream_t stream) {
  (void)n_in; (void)out_size; (void)ws_size;
  const float* pos   = (const float*)d_in[0];
  const float* feat  = (const float*)d_in[1];
  const float* hm    = (const float*)d_in[2];
  const int*   tpl   = (const int*)d_in[3];
  const int*   batch = (const int*)d_in[4];
  const int N = in_sizes[0] / 3;
  const int E = in_sizes[3] / 2;
  const int* src = tpl;
  const int* dst = tpl + E;

  const float* g1p[8]; const float* g2p[8]; const float* g3p[8];
  for (int i = 0; i < 8; ++i) { g1p[i] = (const float*)d_in[5 + i];
                                g2p[i] = (const float*)d_in[13 + i];
                                g3p[i] = (const float*)d_in[21 + i]; }
  const float* m1_w = (const float*)d_in[29]; const float* m1_b = (const float*)d_in[30];
  const float* m1_g = (const float*)d_in[31]; const float* m1_be = (const float*)d_in[32];
  const float* m2_w = (const float*)d_in[33]; const float* m2_b = (const float*)d_in[34];
  const float* m2_g = (const float*)d_in[35]; const float* m2_be = (const float*)d_in[36];
  const float* lin_w = (const float*)d_in[37]; const float* lin_b = (const float*)d_in[38];
  const float* bn_g = (const float*)d_in[39]; const float* bn_b = (const float*)d_in[40];

  // ---- workspace layout ----
  unsigned char* wp = (unsigned char*)d_ws;
  auto alloc = [&](size_t bytes) -> void* {
    void* p = (void*)wp;
    wp += (bytes + 255) & ~(size_t)255;
    return p;
  };
  unsigned short* h1     = (unsigned short*)alloc((size_t)E * 256 * 2);
  int*            agg    = (int*)alloc((size_t)N * 256 * 4);
  unsigned short* xin    = (unsigned short*)alloc((size_t)N * 64 * 2);
  unsigned short* x123bf = (unsigned short*)alloc((size_t)N * 448 * 2);
  unsigned short* h4     = (unsigned short*)alloc((size_t)N * 256 * 2);
  float*          x5     = (float*)alloc((size_t)N * 512 * 4);
  float*          x6raw  = (float*)alloc((size_t)192 * 512 * 4);
  float*          h6     = (float*)alloc((size_t)192 * 256 * 4);
  int*            xgenc  = (int*)alloc((size_t)8 * 448 * 4);
  float*          xgdec  = (float*)alloc((size_t)8 * 448 * 4);
  float*          xg_h   = (float*)alloc((size_t)8 * 256 * 4);
  float*          xgbn   = (float*)alloc((size_t)8 * 256 * 4);
  unsigned short* wrepA  = (unsigned short*)alloc((size_t)448 * 256 * 2);
  unsigned short* wrepB  = (unsigned short*)alloc((size_t)256 * 256 * 2);
  float*          gsum   = (float*)alloc(256 * 4);
  float*          gsq    = (float*)alloc(256 * 4);
  float*          sbuf   = (float*)alloc(256 * 4);
  float*          tbuf   = (float*)alloc(256 * 4);
  float*          s4     = (float*)alloc(256 * 4);
  float*          t4     = (float*)alloc(256 * 4);
  float*          b2p    = (float*)alloc(256 * 4);
  int*            gs     = (int*)alloc(8 * 4);
  int*            ge     = (int*)alloc(8 * 4);

  // ---- x_in (bf16) ----
  build_xin_kernel<<<cdiv_l((long)N * 64, 256), 256, 0, stream>>>(pos, feat, xin, N);

  // ---- GCU layers (WMMA bf16; BN1 folded into W2; seg-max fused into GEMM2) ----
  run_gcu<64, 64>(stream, xin, 64, 0, src, dst, E, N,
                  g1p[0], g1p[1], g1p[2], g1p[3], g1p[4], g1p[5], g1p[6], g1p[7],
                  wrepA, wrepB, h1, agg, gsum, gsq, sbuf, tbuf, b2p, x123bf, 0);
  run_gcu<64, 128>(stream, x123bf, 448, 0, src, dst, E, N,
                   g2p[0], g2p[1], g2p[2], g2p[3], g2p[4], g2p[5], g2p[6], g2p[7],
                   wrepA, wrepB, h1, agg, gsum, gsq, sbuf, tbuf, b2p, x123bf, 64);
  run_gcu<128, 256>(stream, x123bf, 448, 64, src, dst, E, N,
                    g3p[0], g3p[1], g3p[2], g3p[3], g3p[4], g3p[5], g3p[6], g3p[7],
                    wrepA, wrepB, h1, agg, gsum, gsq, sbuf, tbuf, b2p, x123bf, 192);

  // ---- m1: x4 = BN(relu(x123 @ m1_w + b)) ; keep relu output, BN applied at x5 ----
  fill_f32_kernel<<<1, 256, 0, stream>>>(gsum, 0.f, 256);
  fill_f32_kernel<<<1, 256, 0, stream>>>(gsq, 0.f, 256);
  repack_w_kernel<<<cdiv_l((long)448 * 256, 256), 256, 0, stream>>>(m1_w, nullptr, wrepA, 448, 256);
  {
    const int blocksN = cdiv_l(cdiv_l(N, 16), 4);
    gemm_rows_kernel<448, 256, false><<<blocksN, 128, 0, stream>>>(
        x123bf, 448, wrepA, m1_b, nullptr, nullptr, h4, gsum, gsq, N);
  }
  stats_st_kernel<<<1, 256, 0, stream>>>(gsum, gsq, m1_g, m1_be, (float)N, s4, t4, 256);

  // ---- graph embedding: segment-max(x123) -> m2 MLP -> BN ----
  fill_i32_kernel<<<cdiv_l(8 * 448, 256), 256, 0, stream>>>(xgenc, (int)0x80000000, 8 * 448);
  segmax_kernel<<<cdiv_l((long)N * 448, 256), 256, 0, stream>>>(x123bf, batch, xgenc, N);
  xg_decode_kernel<<<cdiv_l(8 * 448, 256), 256, 0, stream>>>(xgenc, xgdec, 8 * 448);
  m2_kernel<<<8, 256, 0, stream>>>(xgdec, m2_w, m2_b, xg_h);
  m2_bn_kernel<<<1, 256, 0, stream>>>(xg_h, m2_g, m2_be, xgbn);

  // ---- x5 = concat(BN(h4), xg_bn[batch]) ----
  x5_kernel<<<cdiv_l((long)N * 512, 256), 256, 0, stream>>>(h4, s4, t4, xgbn, batch, x5, N);

  // ---- einsum over graphs ----
  ranges_init_kernel<<<1, 8, 0, stream>>>(gs, ge, N);
  ranges_kernel<<<cdiv_l(N, 256), 256, 0, stream>>>(batch, gs, ge, N);
  {
    dim3 grid(8, 4);
    einsum_kernel<<<grid, 128, 0, stream>>>(x5, hm, gs, ge, x6raw);
  }

  // ---- final linear + BN over (B,K) ----
  fill_f32_kernel<<<1, 256, 0, stream>>>(gsum, 0.f, 256);
  fill_f32_kernel<<<1, 256, 0, stream>>>(gsq, 0.f, 256);
  lin_kernel<<<192, 256, 0, stream>>>(x6raw, lin_w, lin_b, h6, gsum, gsq);
  stats_st_kernel<<<1, 256, 0, stream>>>(gsum, gsq, bn_g, bn_b, 192.f, sbuf, tbuf, 256);
  final_kernel<<<cdiv_l(192 * 256, 256), 256, 0, stream>>>(h6, sbuf, tbuf, (float*)d_out);
}
